// MHA_13511967113916
// MI455X (gfx1250) — matrix operand order repaired and checked by reading, not compiled, not run on hardware
//
#include <hip/hip_runtime.h>

#define DEV __device__ __forceinline__

typedef __bf16 bf16;
typedef __attribute__((ext_vector_type(16))) __bf16 bf16x16;
typedef __attribute__((ext_vector_type(8)))  __bf16 bf16x8;
typedef __attribute__((ext_vector_type(8)))  float  f32x8;

union BF16X16 { bf16x16 v; bf16x8 h[2]; };

// ---------------- constants (B=2, S=2048, E=2048, H=16, D=128) --------------
constexpr int Bx     = 2;
constexpr int S      = 2048;
constexpr int E      = 2048;
constexpr int H      = 16;
constexpr int D      = 128;
constexpr int M_ROWS = Bx * S;   // 4096
constexpr int QKV_N  = 3 * E;    // 6144

// ---------------- bf16 helpers ---------------------------------------------
DEV bf16 f2bf(float f) {
  unsigned int u = __builtin_bit_cast(unsigned int, f);
  unsigned int r = u + 0x7FFFu + ((u >> 16) & 1u);
  unsigned short s = (unsigned short)(r >> 16);
  return __builtin_bit_cast(bf16, s);
}
DEV float bf2f(bf16 b) {
  unsigned short s = __builtin_bit_cast(unsigned short, b);
  unsigned int u = ((unsigned int)s) << 16;
  return __builtin_bit_cast(float, u);
}
DEV f32x8 zero8() { f32x8 z = {0.f,0.f,0.f,0.f,0.f,0.f,0.f,0.f}; return z; }

// ---------------- conversion kernels ---------------------------------------
__global__ void f32_to_bf16_kernel(const float* __restrict__ in,
                                   bf16* __restrict__ out, int n) {
  int i = blockIdx.x * blockDim.x + threadIdx.x;
  if (i < n) out[i] = f2bf(in[i]);
}

// W[K,N] row-major -> WT[N,K] bf16 (so B fragments load contiguously)
__global__ void transpose_to_bf16_kernel(const float* __restrict__ W,
                                         bf16* __restrict__ WT, int K, int N) {
  int i = blockIdx.x * blockDim.x + threadIdx.x;
  if (i < K * N) {
    int k = i / N, n = i - k * N;
    WT[(size_t)n * K + k] = f2bf(W[i]);
  }
}

// ---------------- GEMM core: one wave computes C[32 x 128] ------------------
// A: row-major [M,K] bf16 (K mult of 32); BT: [N,K] bf16 (B pre-transposed).
// Two 16-row A fragments share every B fragment -> 16 WMMAs per 20 b128 loads.
DEV void wave_gemm_32x128(const bf16* __restrict__ A, const bf16* __restrict__ BT,
                          int K, int mbase, int nbase, int lane,
                          f32x8 acc0[8], f32x8 acc1[8]) {
  const bf16* arow0 = A + (size_t)(mbase + (lane & 15)) * K + ((lane & 16) ? 16 : 0);
  const bf16* arow1 = arow0 + (size_t)16 * K;
  for (int kc = 0; kc < K; kc += 32) {
    BF16X16 a0, a1;
    a0.h[0] = *(const bf16x8*)(arow0 + kc);
    a0.h[1] = *(const bf16x8*)(arow0 + kc + 8);
    a1.h[0] = *(const bf16x8*)(arow1 + kc);
    a1.h[1] = *(const bf16x8*)(arow1 + kc + 8);
    __builtin_prefetch(arow0 + kc + 32, 0, 3);
    __builtin_prefetch(arow1 + kc + 32, 0, 3);
#pragma unroll
    for (int nt = 0; nt < 8; nt++) {
      const bf16* bp = BT + (size_t)(nbase + nt * 16 + (lane & 15)) * K
                          + kc + ((lane & 16) ? 16 : 0);
      bf16x16 b = *(const bf16x16*)bp;
      acc0[nt] = __builtin_amdgcn_wmma_f32_16x16x32_bf16(
          false, a0.v, false, b, (short)0, acc0[nt], false, false);
      acc1[nt] = __builtin_amdgcn_wmma_f32_16x16x32_bf16(
          false, a1.v, false, b, (short)0, acc1[nt], false, false);
    }
  }
}

// ---------------- GEMM1: qkv = x @ Wqkv, scatter into q/k/v [B,H,S,D] ------
__global__ __launch_bounds__(256) void gemm_qkv_kernel(
    const bf16* __restrict__ A, const bf16* __restrict__ BT,
    bf16* __restrict__ qb, bf16* __restrict__ kb, bf16* __restrict__ vb) {
  const int lane = threadIdx.x & 31;
  const int wv   = threadIdx.x >> 5;
  const int mbase = blockIdx.y * 256 + wv * 32;
  const int nbase = blockIdx.x * 128;
  f32x8 acc0[8], acc1[8];
#pragma unroll
  for (int i = 0; i < 8; i++) { acc0[i] = zero8(); acc1[i] = zero8(); }
  wave_gemm_32x128(A, BT, E, mbase, nbase, lane, acc0, acc1);

  const int rowadd = (lane & 16) ? 8 : 0;
  bf16* dsts[3] = {qb, kb, vb};
#pragma unroll
  for (int nt = 0; nt < 8; nt++) {
    int n = nbase + nt * 16 + (lane & 15);
    int which = n >> 11;          // 0=q,1=k,2=v
    int e = n & (E - 1);
    int h = e >> 7, d = e & (D - 1);
#pragma unroll
    for (int mt = 0; mt < 2; mt++) {
#pragma unroll
      for (int r = 0; r < 8; r++) {
        int m = mbase + mt * 16 + r + rowadd;
        int b = m >> 11, s = m & (S - 1);
        float vacc = mt ? acc1[nt][r] : acc0[nt][r];
        dsts[which][(((size_t)(b * H + h)) * S + s) * D + d] = f2bf(vacc);
      }
    }
  }
}

// ---------------- GEMM2: out = O @ Wo (fp32 result) -------------------------
__global__ __launch_bounds__(256) void gemm_out_kernel(
    const bf16* __restrict__ A, const bf16* __restrict__ BT,
    float* __restrict__ out) {
  const int lane = threadIdx.x & 31;
  const int wv   = threadIdx.x >> 5;
  const int mbase = blockIdx.y * 256 + wv * 32;
  const int nbase = blockIdx.x * 128;
  f32x8 acc0[8], acc1[8];
#pragma unroll
  for (int i = 0; i < 8; i++) { acc0[i] = zero8(); acc1[i] = zero8(); }
  wave_gemm_32x128(A, BT, E, mbase, nbase, lane, acc0, acc1);

  const int rowadd = (lane & 16) ? 8 : 0;
#pragma unroll
  for (int nt = 0; nt < 8; nt++) {
    int n = nbase + nt * 16 + (lane & 15);
#pragma unroll
    for (int mt = 0; mt < 2; mt++) {
#pragma unroll
      for (int r = 0; r < 8; r++) {
        int m = mbase + mt * 16 + r + rowadd;
        out[(size_t)m * E + n] = mt ? acc1[nt][r] : acc0[nt][r];
      }
    }
  }
}

// ---------------- RoPE (in-place on q,k) + V transpose ----------------------
__global__ void rope_transpose_kernel(bf16* __restrict__ qb, bf16* __restrict__ kb,
                                      const bf16* __restrict__ vb,
                                      bf16* __restrict__ vT) {
  int idx = blockIdx.x * blockDim.x + threadIdx.x;  // B*H*S*(D/2)
  int d2 = idx & 63;
  int s  = (idx >> 6) & (S - 1);
  int bh = idx >> 17;
  size_t base = ((size_t)bh * S + s) * D + 2 * d2;
  // theta_i = s * 10000^(-2*i/D)
  float invf = __expf((float)d2 * (-2.0f / (float)D) * 9.210340371976184f);
  float th = (float)s * invf;
  float c = __cosf(th), sn = __sinf(th);

  float x0 = bf2f(qb[base]), x1 = bf2f(qb[base + 1]);
  qb[base]     = f2bf(x0 * c - x1 * sn);
  qb[base + 1] = f2bf(x0 * sn + x1 * c);

  x0 = bf2f(kb[base]); x1 = bf2f(kb[base + 1]);
  kb[base]     = f2bf(x0 * c - x1 * sn);
  kb[base + 1] = f2bf(x0 * sn + x1 * c);

  size_t tb = ((size_t)bh * D + 2 * d2) * S + s;   // vT [B,H,D,S]
  vT[tb]     = vb[base];
  vT[tb + S] = vb[base + 1];
}

// ---------------- Flash attention (causal), wave = 16 queries ---------------
__global__ __launch_bounds__(256) void attn_kernel(
    const bf16* __restrict__ qb, const bf16* __restrict__ kbf,
    const bf16* __restrict__ vT, bf16* __restrict__ Ob) {
  __shared__ bf16 Pb[8][16][32];    // per-wave P tile (C-layout -> A-layout)
  const int lane = threadIdx.x & 31;
  const int wv   = threadIdx.x >> 5;
  const int bh   = blockIdx.y;
  const int qbase = blockIdx.x * 128 + wv * 16;
  const int rowadd = (lane & 16) ? 8 : 0;
  const float scale = 0.08838834764831845f;  // 1/sqrt(128)

  const bf16* qhead  = qb  + (size_t)bh * S * D;
  const bf16* khead  = kbf + (size_t)bh * S * D;
  const bf16* vthead = vT  + (size_t)bh * D * S;

  // Q fragments: 16 rows x D, 4 K-chunks of 32, live in registers all loop
  bf16x16 qf[4];
  {
    const bf16* qr = qhead + (size_t)(qbase + (lane & 15)) * D + ((lane & 16) ? 16 : 0);
#pragma unroll
    for (int c = 0; c < 4; c++) {
      BF16X16 u;
      u.h[0] = *(const bf16x8*)(qr + c * 32);
      u.h[1] = *(const bf16x8*)(qr + c * 32 + 8);
      qf[c] = u.v;
    }
  }

  f32x8 o[8];
#pragma unroll
  for (int i = 0; i < 8; i++) o[i] = zero8();
  float m[8], l[8];
#pragma unroll
  for (int r = 0; r < 8; r++) { m[r] = -3.0e38f; l[r] = 0.0f; }

  const int nkt = (qbase + 16 + 31) >> 5;   // causal key-tile count (32 keys/tile)
  for (int kt = 0; kt < nkt; kt++) {
    const int kb32 = kt * 32;

    // ---- scores: two 16x16 C tiles over 32 keys, K=D via 4 chunks ----
    f32x8 sc0 = zero8(), sc1 = zero8();
    {
      const bf16* kr0 = khead + (size_t)(kb32 + (lane & 15)) * D + ((lane & 16) ? 16 : 0);
      const bf16* kr1 = kr0 + (size_t)16 * D;
      __builtin_prefetch(kr0 + (size_t)32 * D, 0, 3);  // next key tile
      __builtin_prefetch(kr1 + (size_t)32 * D, 0, 3);
#pragma unroll
      for (int c = 0; c < 4; c++) {
        bf16x16 b0 = *(const bf16x16*)(kr0 + c * 32);
        bf16x16 b1 = *(const bf16x16*)(kr1 + c * 32);
        sc0 = __builtin_amdgcn_wmma_f32_16x16x32_bf16(false, qf[c], false, b0,
                                                      (short)0, sc0, false, false);
        sc1 = __builtin_amdgcn_wmma_f32_16x16x32_bf16(false, qf[c], false, b1,
                                                      (short)0, sc1, false, false);
      }
    }

    // ---- streaming softmax update (rows split across lane halves) ----
#pragma unroll
    for (int r = 0; r < 8; r++) {
      int row = qbase + r + rowadd;
      int c0  = kb32 + (lane & 15);
      float s0 = sc0[r] * scale; if (c0 > row)      s0 = -3.0e38f;
      float s1 = sc1[r] * scale; if (c0 + 16 > row) s1 = -3.0e38f;
      float mx = fmaxf(s0, s1);
      mx = fmaxf(mx, __shfl_xor(mx, 1, 32));
      mx = fmaxf(mx, __shfl_xor(mx, 2, 32));
      mx = fmaxf(mx, __shfl_xor(mx, 4, 32));
      mx = fmaxf(mx, __shfl_xor(mx, 8, 32));
      float mn = fmaxf(m[r], mx);
      float f  = __expf(m[r] - mn);
      m[r] = mn;
      s0 = __expf(s0 - mn);
      s1 = __expf(s1 - mn);
      float rs = s0 + s1;
      rs += __shfl_xor(rs, 1, 32);
      rs += __shfl_xor(rs, 2, 32);
      rs += __shfl_xor(rs, 4, 32);
      rs += __shfl_xor(rs, 8, 32);
      l[r] = l[r] * f + rs;
#pragma unroll
      for (int dt = 0; dt < 8; dt++) o[dt][r] *= f;
      sc0[r] = s0; sc1[r] = s1;
    }

    // ---- P: C-layout -> LDS -> A-layout fragment ----
#pragma unroll
    for (int r = 0; r < 8; r++) {
      Pb[wv][r + rowadd][lane & 15]        = f2bf(sc0[r]);
      Pb[wv][r + rowadd][16 + (lane & 15)] = f2bf(sc1[r]);
    }
    __builtin_amdgcn_wave_barrier();
    BF16X16 pa;
    {
      const bf16* pr = &Pb[wv][lane & 15][0];
      int kb8 = (lane & 16) ? 16 : 0;
      pa.h[0] = *(const bf16x8*)(pr + kb8);
      pa.h[1] = *(const bf16x8*)(pr + kb8 + 8);
    }
    __builtin_amdgcn_wave_barrier();

    // ---- O += P @ V  (V from transposed [D,S] layout, contiguous K) ----
    {
      const bf16* vr = vthead + (size_t)(lane & 15) * S + kb32 + ((lane & 16) ? 16 : 0);
#pragma unroll
      for (int dt = 0; dt < 8; dt++) {
        bf16x16 bv = *(const bf16x16*)(vr + (size_t)dt * 16 * S);
        o[dt] = __builtin_amdgcn_wmma_f32_16x16x32_bf16(false, pa.v, false, bv,
                                                        (short)0, o[dt], false, false);
      }
    }
  }

  // ---- epilogue: normalize and write O_b [B,S,E] bf16 ----
  const int b = bh >> 4, h = bh & 15;
#pragma unroll
  for (int r = 0; r < 8; r++) {
    int srow = qbase + r + rowadd;
    float inv = 1.0f / l[r];
    bf16* dst = Ob + ((size_t)(b * S + srow)) * E + h * D + (lane & 15);
#pragma unroll
    for (int dt = 0; dt < 8; dt++) dst[dt * 16] = f2bf(o[dt][r] * inv);
  }
}

// ---------------- host-side launcher ----------------------------------------
extern "C" void kernel_launch(void* const* d_in, const int* in_sizes, int n_in,
                              void* d_out, int out_size, void* d_ws, size_t ws_size,
                              hipStream_t stream) {
  (void)in_sizes; (void)n_in; (void)out_size; (void)ws_size;
  const float* x    = (const float*)d_in[0];
  const float* Wqkv = (const float*)d_in[1];
  const float* Wo   = (const float*)d_in[2];
  float* out = (float*)d_out;

  char* p = (char*)d_ws;
  bf16* xb    = (bf16*)p; p += (size_t)M_ROWS * E * 2;       // 16 MB
  bf16* WqkvT = (bf16*)p; p += (size_t)QKV_N * E * 2;        // 24 MB
  bf16* WoT   = (bf16*)p; p += (size_t)E * E * 2;            //  8 MB
  bf16* qbuf  = (bf16*)p; p += (size_t)Bx * H * S * D * 2;   // 16 MB
  bf16* kbuf  = (bf16*)p; p += (size_t)Bx * H * S * D * 2;   // 16 MB
  bf16* vbuf  = (bf16*)p; p += (size_t)Bx * H * S * D * 2;   // 16 MB
  bf16* vT    = (bf16*)p; p += (size_t)Bx * H * D * S * 2;   // 16 MB
  bf16* Ob    = (bf16*)p; p += (size_t)M_ROWS * E * 2;       // 16 MB (128 MB total)

  f32_to_bf16_kernel<<<(M_ROWS * E) / 256, 256, 0, stream>>>(x, xb, M_ROWS * E);
  transpose_to_bf16_kernel<<<(E * QKV_N) / 256, 256, 0, stream>>>(Wqkv, WqkvT, E, QKV_N);
  transpose_to_bf16_kernel<<<(E * E) / 256, 256, 0, stream>>>(Wo, WoT, E, E);

  gemm_qkv_kernel<<<dim3(QKV_N / 128, M_ROWS / 256), 256, 0, stream>>>(
      xb, WqkvT, qbuf, kbuf, vbuf);

  rope_transpose_kernel<<<(Bx * H * S * (D / 2)) / 256, 256, 0, stream>>>(
      qbuf, kbuf, vbuf, vT);

  attn_kernel<<<dim3(S / 128, Bx * H), 256, 0, stream>>>(qbuf, kbuf, vT, Ob);

  gemm_out_kernel<<<dim3(E / 128, M_ROWS / 256), 256, 0, stream>>>(Ob, WoT, out);
}
